// DecoderRNN_48593259987376
// MI455X (gfx1250) — matrix operand order repaired and checked by reading, not compiled, not run on hardware
//
#include <hip/hip_runtime.h>
#include <hip/hip_bf16.h>
#include <math.h>

typedef __attribute__((ext_vector_type(16))) _Float16 v16h;
typedef __attribute__((ext_vector_type(8)))  _Float16 v8h;
typedef __attribute__((ext_vector_type(8)))  float    v8f;

// ---------------------------------------------------------------------------
// WMMA fragment loaders (CDNA5 16x16x32 f16 layouts, ISA 7.12.2)
// A: 16(M) x 32(K), memory row-major [M,K] (K contiguous), ld = K
//    lane<16 : row=lane,    elems 0..7 = K+0..7,  elems 8..15 = K+16..23
//    lane>=16: row=lane-16, elems 0..7 = K+8..15, elems 8..15 = K+24..31
// ---------------------------------------------------------------------------
__device__ __forceinline__ v16h load_a_frag(const _Float16* __restrict__ base,
                                            int ld, int lane) {
  const int half = lane >> 4;
  const int r    = lane & 15;
  const _Float16* p = base + (long long)r * ld + half * 8;
  v8h lo = *(const v8h*)(p);        // K chunk +0 (or +8)
  v8h hi = *(const v8h*)(p + 16);   // K chunk +16 (or +24)
  v16h v;
#pragma unroll
  for (int i = 0; i < 8; ++i) { v[i] = lo[i]; v[i + 8] = hi[i]; }
  return v;
}

// B: 32(K) x 16(N), weight stored row-major [N,K] (K contiguous), ld = K
//    lane<16 : col=lane,    elems 0..7 = K+0..7,  elems 8..15 = K+16..23
//    lane>=16: col=lane-16, elems 0..7 = K+8..15, elems 8..15 = K+24..31
__device__ __forceinline__ v16h load_b_frag(const _Float16* __restrict__ base,
                                            int ld, int lane) {
  const int half = lane >> 4;
  const int n    = lane & 15;
  const _Float16* p = base + (long long)n * ld + half * 8;
  v8h lo = *(const v8h*)(p);
  v8h hi = *(const v8h*)(p + 16);
  v16h v;
#pragma unroll
  for (int i = 0; i < 8; ++i) { v[i] = lo[i]; v[i + 8] = hi[i]; }
  return v;
}

// ---------------------------------------------------------------------------
// Generic C = act( A*B^T [+ A2*B2^T] + bias ) WMMA GEMM, specialized for
// M == 64 (true for every GEMM in this model: M = batch = 64).
// A:[64,K] f16 row-major. B:[N,K] f16 row-major (classic W with K inner).
// One wave computes a 64x16 strip of C (4 M-tiles x 1 N-tile): the streamed
// B fragment (the large, L2-resident weight operand) is loaded ONCE per
// K-step and reused across 4 WMMAs; the tiny A matrix (64 KB, shared by all
// waves) is re-read from WGP$/L0.  4 independent accumulators keep the
// matrix pipe free of back-to-back RAW chains. 8 waves (256 thr) per block.
// ---------------------------------------------------------------------------
__global__ void k_wmma_gemm(const _Float16* __restrict__ A,
                            const _Float16* __restrict__ B, int K,
                            const _Float16* __restrict__ A2,
                            const _Float16* __restrict__ B2, int K2,
                            const float* __restrict__ bias,
                            float* __restrict__ C, _Float16* __restrict__ Ch,
                            int N, int relu) {
  const int lane = threadIdx.x & 31;
  const int wave = threadIdx.x >> 5;
  const int nt   = blockIdx.x * (blockDim.x >> 5) + wave;  // one N-tile / wave
  if (nt >= (N >> 4)) return;                // wave-uniform; EXEC stays full

  const _Float16* Bb = B + (long long)(nt * 16) * K;
  v8f acc0 = {}, acc1 = {}, acc2 = {}, acc3 = {};
  for (int k = 0; k < K; k += 32) {
    v16h b  = load_b_frag(Bb + k, K, lane);
    v16h a0 = load_a_frag(A + k,                     K, lane);
    v16h a1 = load_a_frag(A + k + (long long)16 * K, K, lane);
    v16h a2 = load_a_frag(A + k + (long long)32 * K, K, lane);
    v16h a3 = load_a_frag(A + k + (long long)48 * K, K, lane);
    acc0 = __builtin_amdgcn_wmma_f32_16x16x32_f16(false, a0, false, b,
                                                  (short)0, acc0, false, false);
    acc1 = __builtin_amdgcn_wmma_f32_16x16x32_f16(false, a1, false, b,
                                                  (short)0, acc1, false, false);
    acc2 = __builtin_amdgcn_wmma_f32_16x16x32_f16(false, a2, false, b,
                                                  (short)0, acc2, false, false);
    acc3 = __builtin_amdgcn_wmma_f32_16x16x32_f16(false, a3, false, b,
                                                  (short)0, acc3, false, false);
  }
  if (A2) {
    const _Float16* B2b = B2 + (long long)(nt * 16) * K2;
    for (int k = 0; k < K2; k += 32) {
      v16h b  = load_b_frag(B2b + k, K2, lane);
      v16h a0 = load_a_frag(A2 + k,                      K2, lane);
      v16h a1 = load_a_frag(A2 + k + (long long)16 * K2, K2, lane);
      v16h a2 = load_a_frag(A2 + k + (long long)32 * K2, K2, lane);
      v16h a3 = load_a_frag(A2 + k + (long long)48 * K2, K2, lane);
      acc0 = __builtin_amdgcn_wmma_f32_16x16x32_f16(false, a0, false, b,
                                                    (short)0, acc0, false, false);
      acc1 = __builtin_amdgcn_wmma_f32_16x16x32_f16(false, a1, false, b,
                                                    (short)0, acc1, false, false);
      acc2 = __builtin_amdgcn_wmma_f32_16x16x32_f16(false, a2, false, b,
                                                    (short)0, acc2, false, false);
      acc3 = __builtin_amdgcn_wmma_f32_16x16x32_f16(false, a3, false, b,
                                                    (short)0, acc3, false, false);
    }
  }
  // C/D layout: lane -> (half = lane>>4, col = lane&15); acc[j] -> row half*8+j
  const int half = lane >> 4;
  const int n    = nt * 16 + (lane & 15);
  const float bv = bias ? bias[n] : 0.0f;    // loaded once, reused 4 tiles
  v8f accs[4] = {acc0, acc1, acc2, acc3};
#pragma unroll
  for (int mt = 0; mt < 4; ++mt) {
#pragma unroll
    for (int j = 0; j < 8; ++j) {
      const int m = mt * 16 + half * 8 + j;
      float v = accs[mt][j] + bv;
      if (relu) v = v > 0.0f ? v : 0.0f;
      if (C)  C[(long long)m * N + n] = v;
      if (Ch) Ch[(long long)m * N + n] = (_Float16)v;
    }
  }
}

// ---------------------------------------------------------------------------
// small helpers
// ---------------------------------------------------------------------------
__global__ void k_f32_to_f16(const float* __restrict__ s,
                             _Float16* __restrict__ d, int n) {
  int i = blockIdx.x * blockDim.x + threadIdx.x;
  if (i < n) d[i] = (_Float16)s[i];
}

__global__ void k_add_f32(const float* __restrict__ a,
                          const float* __restrict__ b,
                          float* __restrict__ o, int n) {
  int i = blockIdx.x * blockDim.x + threadIdx.x;
  if (i < n) o[i] = a[i] + b[i];
}

__global__ void k_copy_f32(const float* __restrict__ s,
                           float* __restrict__ d, int n) {
  int i = blockIdx.x * blockDim.x + threadIdx.x;
  if (i < n) d[i] = s[i];
}

__global__ void k_init_state(const float* __restrict__ h_in,
                             const float* __restrict__ c_in,
                             float* __restrict__ h32, float* __restrict__ c32,
                             _Float16* __restrict__ h16, int n) {
  int i = blockIdx.x * blockDim.x + threadIdx.x;
  if (i < n) {
    float hv = h_in[i];
    h32[i] = hv; h16[i] = (_Float16)hv; c32[i] = c_in[i];
  }
}

// builds a_in = [e, h_top] (f16) and the e-half of xc = [e, context]
__global__ void k_embed(const int* __restrict__ cur,
                        const int* __restrict__ gtruth, int t,
                        const float* __restrict__ emb,
                        const float* __restrict__ h32_top,
                        _Float16* __restrict__ a_in, _Float16* __restrict__ xc,
                        int B, int E, int H) {
  int i = blockIdx.x * blockDim.x + threadIdx.x;
  const int W = E + H;
  if (i >= B * W) return;
  int b = i / W, j = i % W;
  if (j < E) {
    int tok = (t == 0) ? cur[b] : gtruth[(t - 1) * B + b];
    _Float16 v = (_Float16)emb[(long long)tok * E + j];
    a_in[i] = v;
    xc[i]   = v;
  } else {
    a_in[i] = (_Float16)h32_top[b * H + (j - E)];
  }
}

// softmax over MAXLEN logits, then context = attn_w[:S] . enc; fills xc[:,E:]
__global__ void k_softmax_ctx(const float* __restrict__ logits, // [B,ML]
                              const float* __restrict__ enc,    // [S,B,H]
                              _Float16* __restrict__ xc,        // [B,E+H]
                              int B, int S, int H, int E, int ML) {
  __shared__ float w[64];
  const int b = blockIdx.x, tid = threadIdx.x;
  if (tid < ML) w[tid] = logits[b * ML + tid];
  __syncthreads();
  if (tid == 0) {
    float mx = w[0];
    for (int i = 1; i < ML; ++i) mx = fmaxf(mx, w[i]);
    float sum = 0.0f;
    for (int i = 0; i < ML; ++i) { w[i] = expf(w[i] - mx); sum += w[i]; }
    float inv = 1.0f / sum;
    for (int i = 0; i < ML; ++i) w[i] *= inv;
  }
  __syncthreads();
  for (int h = tid; h < H; h += blockDim.x) {
    float s = 0.0f;
    for (int si = 0; si < S; ++si)
      s += w[si] * enc[((long long)si * B + b) * H + h];
    xc[b * (E + H) + E + h] = (_Float16)s;
  }
}

// gates [B,4H] (i,f,g,o) -> updates c, h (f32 + f16 mirror)
__global__ void k_lstm_elem(const float* __restrict__ g, float* __restrict__ c,
                            float* __restrict__ h32, _Float16* __restrict__ h16,
                            int B, int H) {
  int i = blockIdx.x * blockDim.x + threadIdx.x;
  if (i >= B * H) return;
  int b = i / H, hh = i % H;
  const float* gb = g + (long long)b * 4 * H;
  float ig = 1.0f / (1.0f + expf(-gb[hh]));
  float fg = 1.0f / (1.0f + expf(-gb[H + hh]));
  float gg = tanhf(gb[2 * H + hh]);
  float og = 1.0f / (1.0f + expf(-gb[3 * H + hh]));
  float cn = fg * c[i] + ig * gg;
  float hn = og * tanhf(cn);
  c[i] = cn; h32[i] = hn; h16[i] = (_Float16)hn;
}

// in-place log-softmax over one row of length Vn per block
__global__ void k_logsoftmax(float* __restrict__ base, int Vn) {
  __shared__ float red[256];
  float* row = base + (long long)blockIdx.x * Vn;
  const int tid = threadIdx.x;
  float mx = -INFINITY;
  for (int i = tid; i < Vn; i += blockDim.x) mx = fmaxf(mx, row[i]);
  red[tid] = mx; __syncthreads();
  for (int s = blockDim.x >> 1; s > 0; s >>= 1) {
    if (tid < s) red[tid] = fmaxf(red[tid], red[tid + s]);
    __syncthreads();
  }
  mx = red[0]; __syncthreads();
  float sum = 0.0f;
  for (int i = tid; i < Vn; i += blockDim.x) sum += expf(row[i] - mx);
  red[tid] = sum; __syncthreads();
  for (int s = blockDim.x >> 1; s > 0; s >>= 1) {
    if (tid < s) red[tid] += red[tid + s];
    __syncthreads();
  }
  const float lse = mx + logf(red[0]);
  __syncthreads();
  for (int i = tid; i < Vn; i += blockDim.x) row[i] -= lse;
}

// ---------------------------------------------------------------------------
extern "C" void kernel_launch(void* const* d_in, const int* in_sizes, int n_in,
                              void* d_out, int out_size, void* d_ws, size_t ws_size,
                              hipStream_t stream) {
  const int*   cur      = (const int*)d_in[0];
  const float* hidden_h = (const float*)d_in[1];
  const float* hidden_c = (const float*)d_in[2];
  /* d_in[3] = length (device scalar; T recovered from gtruth shape) */
  const float* enc      = (const float*)d_in[4];
  const int*   gtruth   = (const int*)d_in[5];
  const float* emb      = (const float*)d_in[6];
  const float* W_attn   = (const float*)d_in[7];
  const float* W_ao     = (const float*)d_in[8];
  const float* W_ih     = (const float*)d_in[9];
  const float* W_hh     = (const float*)d_in[10];
  const float* b_ih     = (const float*)d_in[11];
  const float* b_hh     = (const float*)d_in[12];
  const float* W_gen    = (const float*)d_in[13];
  const float* b_gen    = (const float*)d_in[14];
  float* out = (float*)d_out;

  const int B = 64, E = 512, H = 512, L = 3, ML = 64, V = 32000;
  const int S = in_sizes[4] / (B * H);
  const int T = in_sizes[5] / B;

  // ---- workspace carve (256B aligned) ----
  char* p = (char*)d_ws;
  auto carve = [&](size_t bytes) -> void* {
    void* r = (void*)p;
    p += (bytes + 255) & ~(size_t)255;
    return r;
  };
  _Float16* Wattn_h = (_Float16*)carve((size_t)ML * (E + H) * 2);
  _Float16* Wao_h   = (_Float16*)carve((size_t)E * (E + H) * 2);
  _Float16* Wih_h   = (_Float16*)carve((size_t)L * 4 * H * E * 2);
  _Float16* Whh_h   = (_Float16*)carve((size_t)L * 4 * H * H * 2);
  _Float16* Wgen_h  = (_Float16*)carve((size_t)V * H * 2);
  float*    bcomb   = (float*)   carve((size_t)L * 4 * H * 4);
  float*    h32     = (float*)   carve((size_t)L * B * H * 4);
  float*    c32     = (float*)   carve((size_t)L * B * H * 4);
  _Float16* h16     = (_Float16*)carve((size_t)L * B * H * 2);
  _Float16* a_in    = (_Float16*)carve((size_t)B * (E + H) * 2);
  _Float16* xc      = (_Float16*)carve((size_t)B * (E + H) * 2);
  _Float16* x_h     = (_Float16*)carve((size_t)B * E * 2);
  float*    attnL   = (float*)   carve((size_t)B * ML * 4);
  float*    gates   = (float*)   carve((size_t)B * 4 * H * 4);

  auto cvt = [&](const float* s, _Float16* d, int n) {
    k_f32_to_f16<<<(n + 255) / 256, 256, 0, stream>>>(s, d, n);
  };
  // ---- one-time (per launch) weight conversion: HBM -> L2-resident f16 ----
  cvt(W_attn, Wattn_h, ML * (E + H));
  cvt(W_ao,   Wao_h,   E * (E + H));
  cvt(W_ih,   Wih_h,   L * 4 * H * E);
  cvt(W_hh,   Whh_h,   L * 4 * H * H);
  cvt(W_gen,  Wgen_h,  V * H);
  k_add_f32<<<(L * 4 * H + 255) / 256, 256, 0, stream>>>(b_ih, b_hh, bcomb, L * 4 * H);
  k_init_state<<<(L * B * H + 255) / 256, 256, 0, stream>>>(
      hidden_h, hidden_c, h32, c32, h16, L * B * H);

  auto gemm = [&](const _Float16* A, const _Float16* Bm, int K,
                  const _Float16* A2, const _Float16* B2, int K2,
                  const float* bias, float* C, _Float16* Ch,
                  int N, int relu) {
    int jobs   = N / 16;                    // one wave per 64x16 strip (M=64)
    int blocks = (jobs + 7) / 8;            // 8 waves / block
    k_wmma_gemm<<<blocks, 256, 0, stream>>>(A, Bm, K, A2, B2, K2, bias, C, Ch,
                                            N, relu);
  };

  for (int t = 0; t < T; ++t) {
    // token embed + concat with top-layer hidden
    k_embed<<<(B * (E + H) + 255) / 256, 256, 0, stream>>>(
        cur, gtruth, t, emb, h32 + 2 * B * H, a_in, xc, B, E, H);
    // attention logits: [B, E+H] x [ML, E+H]^T
    gemm(a_in, Wattn_h, E + H, nullptr, nullptr, 0, nullptr, attnL, nullptr,
         ML, 0);
    // softmax + context -> xc[:, E:]
    k_softmax_ctx<<<B, 128, 0, stream>>>(attnL, enc, xc, B, S, H, E, ML);
    // x = relu(xc @ W_attn_out^T)  -> f16
    gemm(xc, Wao_h, E + H, nullptr, nullptr, 0, nullptr, nullptr, x_h,
         E, 1);
    // 3 LSTM layers: gates = x@W_ih^T + h_l@W_hh^T + (b_ih+b_hh)
    for (int l = 0; l < L; ++l) {
      const _Float16* xin = (l == 0) ? x_h : (h16 + (l - 1) * B * H);
      gemm(xin, Wih_h + (size_t)l * 4 * H * E, E,
           h16 + l * B * H, Whh_h + (size_t)l * 4 * H * H, H,
           bcomb + l * 4 * H, gates, nullptr, 4 * H, 0);
      k_lstm_elem<<<(B * H + 255) / 256, 256, 0, stream>>>(
          gates, c32 + l * B * H, h32 + l * B * H, h16 + l * B * H, B, H);
    }
    // vocab logits straight into d_out, then in-place log-softmax
    float* step_out = out + (long long)t * B * V;
    gemm(h16 + 2 * B * H, Wgen_h, H, nullptr, nullptr, 0, b_gen, step_out,
         nullptr, V, 0);
    k_logsoftmax<<<B, 256, 0, stream>>>(step_out, V);
  }

  // final hidden / cell states appended after outputs
  float* hout = out + (long long)T * B * V;
  k_copy_f32<<<(L * B * H + 255) / 256, 256, 0, stream>>>(h32, hout, L * B * H);
  k_copy_f32<<<(L * B * H + 255) / 256, 256, 0, stream>>>(c32, hout + L * B * H,
                                                          L * B * H);
}
